// RegionalGNN_40269613367501
// MI455X (gfx1250) — matrix operand order repaired and checked by reading, not compiled, not run on hardware
//
#include <hip/hip_runtime.h>
#include <hip/hip_bf16.h>
#include <math.h>

typedef _Float16 half_t;
typedef __attribute__((ext_vector_type(16))) _Float16 v16h;
typedef __attribute__((ext_vector_type(8)))  _Float16 v8h;
typedef __attribute__((ext_vector_type(8)))  float    v8f;

#define NB    8192      // batch
#define DIN   512
#define HH    256       // hidden
#define HD    1024      // qkv width
#define NLAY  3
#define NROWS 65536     // NB * 8 countries

// adjacency rows as bitmasks (bit j set => edge i->j)
__constant__ unsigned char c_adj[8] = {0x7E,0x85,0x2B,0x75,0x69,0x1D,0x19,0x02};
__constant__ float c_gdp[8] = {0.40f,0.15f,0.12f,0.10f,0.08f,0.08f,0.05f,0.02f};

union U16 { v16h v; v8h h[2]; };

// ---- WMMA fragment loaders (16-bit, 16x16x32, wave32; A and B share a contiguous k order) ----
// A (16x32): lane = half*16 + row(0..15); elem i -> K = k0 + half*16 + i (contiguous)
__device__ __forceinline__ v16h frag_a(const half_t* base, int ld, int row0, int k0, int lane) {
  int r  = row0 + (lane & 15);
  int hf = lane >> 4;
  U16 u;
  u.h[0] = *(const v8h*)(base + r * ld + k0 + hf * 16);
  u.h[1] = *(const v8h*)(base + r * ld + k0 + hf * 16 + 8);
  return u.v;
}
// B (32x16) from transposed weights Wt[N][K]: lane = half*16 + col; elem i -> K = k0 + half*16 + i (contiguous)
__device__ __forceinline__ v16h frag_b(const half_t* wt, int ld, int col0, int k0, int lane) {
  int c  = col0 + (lane & 15);
  int hf = lane >> 4;
  return *(const v16h*)(wt + c * ld + k0 + hf * 16);
}

// ---------------- elementwise converts ----------------
__global__ void k_cvt_f16(const float* __restrict__ in, half_t* __restrict__ out, int n) {
  int i = blockIdx.x * blockDim.x + threadIdx.x;
  if (i < n) out[i] = (half_t)in[i];
}
// in: [K][N] f32 row-major  -> out: [N][K] f16 (transposed)
__global__ void k_transpose_f16(const float* __restrict__ in, half_t* __restrict__ out, int K, int N) {
  int idx = blockIdx.x * blockDim.x + threadIdx.x;
  if (idx >= K * N) return;
  int n = idx / K, k = idx - n * K;
  out[idx] = (half_t)in[k * N + n];
}

// ---------------- generic WMMA GEMM: C_f32[M,N] = A_f16[M,K] @ Bt_f16[N,K]^T + bias ----------------
// block = 256 thr (8 waves) computes a 32x128 tile; grid = (M/32, N/128)
__global__ __launch_bounds__(256)
void k_wmma_gemm_bias(const half_t* __restrict__ A, const half_t* __restrict__ Bt,
                      const float* __restrict__ bias, float* __restrict__ C,
                      int M, int N, int K) {
  int row0 = blockIdx.x * 32;
  int col0 = blockIdx.y * 128;
  int lane = threadIdx.x & 31, wave = threadIdx.x >> 5;
  int mt = wave >> 2, ntb = (wave & 3) * 2;
  for (int q = 0; q < 2; ++q) {
    int nt = ntb + q;
    v8f acc = {};
    for (int k0 = 0; k0 < K; k0 += 32) {
      v16h a = frag_a(A, K, row0 + mt * 16, k0, lane);
      v16h b = frag_b(Bt, K, col0 + nt * 16, k0, lane);
      acc = __builtin_amdgcn_wmma_f32_16x16x32_f16(false, a, false, b, (short)0, acc, false, false);
    }
    int col = lane & 15, rh = lane >> 4;
    for (int r = 0; r < 8; ++r) {
      int rr = row0 + mt * 16 + rh * 8 + r;
      int cc = col0 + nt * 16 + col;
      C[rr * N + cc] = acc[r] + bias[cc];
    }
  }
}

// ---------------- broadcast over countries + embedding ----------------
__global__ void k_expand(const float* __restrict__ h, const float* __restrict__ emb,
                         float* __restrict__ x32, half_t* __restrict__ x16) {
  int idx = blockIdx.x * 256 + threadIdx.x;          // NROWS*HH total
  int k = idx & 255, row = idx >> 8;
  int b = row >> 3, c = row & 7;
  float v = h[b * HH + k] + emb[c * HH + k];
  x32[idx] = v;
  x16[idx] = (half_t)v;
}

// helper: Out_f16[32][128] (in LDS) = sX[32][256] @ Wt[d0..d0+127][256]^T + bias[d0..]
__device__ __forceinline__ void qkv_chunk(const half_t* sX, half_t* sOut,
                                          const half_t* __restrict__ Wt,
                                          const float* __restrict__ bias,
                                          int d0, int wave, int lane) {
  for (int tt = wave; tt < 16; tt += 8) {            // 2 m-tiles x 8 n-tiles
    int mt = tt >> 3, nt = tt & 7;
    v8f acc = {};
    for (int k0 = 0; k0 < HH; k0 += 32) {
      v16h a = frag_a(sX, HH, mt * 16, k0, lane);
      v16h b = frag_b(Wt, HH, d0 + nt * 16, k0, lane);
      acc = __builtin_amdgcn_wmma_f32_16x16x32_f16(false, a, false, b, (short)0, acc, false, false);
    }
    int col = lane & 15, rh = lane >> 4;
    for (int r = 0; r < 8; ++r) {
      float v = acc[r] + bias[d0 + nt * 16 + col];
      sOut[(mt * 16 + rh * 8 + r) * 128 + nt * 16 + col] = (half_t)v;
    }
  }
}

// ---------------- fused attention layer: 1 block = 4 batches (32 rows) ----------------
__global__ __launch_bounds__(256)
void k_layer(float* __restrict__ x32, half_t* __restrict__ x16,
             const half_t* __restrict__ Wqt, const float* __restrict__ bq,
             const half_t* __restrict__ Wkt, const float* __restrict__ bkv,
             const half_t* __restrict__ Wvt, const float* __restrict__ bv,
             const half_t* __restrict__ Wot, const float* __restrict__ bo,
             const float* __restrict__ lng, const float* __restrict__ lnb) {
  __shared__ __align__(32) unsigned char smem[32 * 1024 + 256 * 4];
  half_t* sX = (half_t*)smem;                 // [32][256] f16, 16KB   (phases 1-4)
  half_t* sQ = (half_t*)(smem + 16384);       // [32][128] f16,  8KB   (Q chunk / V chunk)
  half_t* sK = (half_t*)(smem + 24576);       // [32][128] f16,  8KB   (K chunk / O chunk)
  float*  sY = (float*)smem;                  // [32][256] f32, 32KB   (aliases sX/sQ/sK, phases 5-6)
  float*  sS = (float*)(smem + 32768);        // [4][8][8] scores/attn

  int t = threadIdx.x, lane = t & 31, wave = t >> 5;
  int row0 = blockIdx.x * 32;

  // phase 1: stage x tile (f16) into LDS, zero scores
  for (int i = t; i < 32 * 256 / 8; i += 256)
    ((v8h*)sX)[i] = ((const v8h*)(x16 + row0 * HH))[i];
  sS[t] = 0.0f;
  // prefetch first weight chunks into L2/L1 (global_prefetch_b8)
  __builtin_prefetch(Wqt + (size_t)t * 128, 0, 3);
  __builtin_prefetch(Wkt + (size_t)t * 128, 0, 3);
  __syncthreads();

  // phase 2: scores, accumulated over 128-wide d-chunks (Q,K never hit HBM)
  for (int c = 0; c < 8; ++c) {
    int d0 = c * 128;
    if (c < 7) {  // prefetch next chunk's weights (64KB each) while this chunk computes
      __builtin_prefetch(Wqt + (size_t)(d0 + 128) * HH + (size_t)t * 128, 0, 3);
      __builtin_prefetch(Wkt + (size_t)(d0 + 128) * HH + (size_t)t * 128, 0, 3);
    }
    qkv_chunk(sX, sQ, Wqt, bq, d0, wave, lane);
    qkv_chunk(sX, sK, Wkt, bkv, d0, wave, lane);
    __syncthreads();
    {
      int b = t >> 6, i = (t >> 3) & 7, j = t & 7;    // t == b*64 + i*8 + j
      const v8h* qv = (const v8h*)(sQ + (b * 8 + i) * 128);
      const v8h* kv = (const v8h*)(sK + (b * 8 + j) * 128);
      float p = 0.0f;
      for (int d = 0; d < 16; ++d) {                  // 16 x b128 LDS loads each
        v8h qa = qv[d], ka = kv[d];
        for (int e = 0; e < 8; ++e) p += (float)qa[e] * (float)ka[e];
      }
      sS[t] += p;
    }
    __syncthreads();
  }

  // phase 3: masked softmax (scale = 1/sqrt(256) = 0.0625)
  if (t < 32) {
    int b = t >> 3, i = t & 7;
    float* srow = sS + b * 64 + i * 8;
    unsigned m = c_adj[i];
    float mx = -1e30f;
    for (int j = 0; j < 8; ++j)
      if (m & (1u << j)) { float v = srow[j] * 0.0625f; srow[j] = v; if (v > mx) mx = v; }
    float sum = 0.0f;
    for (int j = 0; j < 8; ++j) {
      float e = (m & (1u << j)) ? __expf(srow[j] - mx) : 0.0f;
      srow[j] = e; sum += e;
    }
    float inv = 1.0f / sum;
    for (int j = 0; j < 8; ++j) srow[j] *= inv;
  }
  __syncthreads();

  // phase 4: y[32,256] = (attn @ V) @ Wo, chunk-wise; per-wave f32 accumulators (2x16 tiles)
  v8f acc[4] = {};
  int mt = wave >> 2, ntb = (wave & 3) * 4;
  for (int c = 0; c < 8; ++c) {
    int d0 = c * 128;
    if (c < 7) {
      __builtin_prefetch(Wvt + (size_t)(d0 + 128) * HH + (size_t)t * 128, 0, 3);
      __builtin_prefetch(Wot + (size_t)t * HD + d0 + 128, 0, 3);   // next k-slice of each Wo row
    }
    qkv_chunk(sX, sQ, Wvt, bv, d0, wave, lane);      // V chunk -> sQ
    __syncthreads();
    {                                                // O chunk = attn @ Vc -> sK (vectorized)
      int row = t >> 3, cb = (t & 7) * 16;
      int b = row >> 3;
      const float* arow = sS + b * 64 + (row & 7) * 8;
      float o[16] = {};
      for (int j = 0; j < 8; ++j) {
        float aw = arow[j];
        const v8h* vv = (const v8h*)(sQ + (b * 8 + j) * 128 + cb);
        v8h v0 = vv[0], v1 = vv[1];
        for (int e = 0; e < 8; ++e) { o[e] += aw * (float)v0[e]; o[8 + e] += aw * (float)v1[e]; }
      }
      v8h r0, r1;
      for (int e = 0; e < 8; ++e) { r0[e] = (half_t)o[e]; r1[e] = (half_t)o[8 + e]; }
      v8h* dst = (v8h*)(sK + row * 128 + cb);
      dst[0] = r0; dst[1] = r1;
    }
    __syncthreads();
    for (int k0 = 0; k0 < 128; k0 += 32) {           // y += Oc @ Wo[d0+k, :]
      v16h a = frag_a(sK, 128, mt * 16, k0, lane);
      for (int q = 0; q < 4; ++q) {
        v16h b = frag_b(Wot, HD, (ntb + q) * 16, d0 + k0, lane);
        acc[q] = __builtin_amdgcn_wmma_f32_16x16x32_f16(false, a, false, b, (short)0, acc[q], false, false);
      }
    }
    __syncthreads();
  }

  // phase 5: spill accumulators (+bo) to sY (aliases sX/sQ/sK — all dead now)
  {
    int col = lane & 15, rh = lane >> 4;
    for (int q = 0; q < 4; ++q)
      for (int r = 0; r < 8; ++r) {
        int rr = mt * 16 + rh * 8 + r, cc = (ntb + q) * 16 + col;
        sY[rr * HH + cc] = acc[q][r] + bo[cc];
      }
  }
  __syncthreads();

  // phase 6: residual + layernorm, write x back. 8 lanes per row, wave32 shfl_xor reductions.
  {
    int row = t >> 3;                                // 0..31 (4 rows per wave)
    int sub = t & 7;                                 // lane group of 8 within the wave
    int rowg = row0 + row;
    float* yr = sY + row * HH;
    float s = 0.0f;
    for (int k = sub; k < HH; k += 8) {              // pass 1: residual add + sum
      float v = x32[rowg * HH + k] + yr[k];
      yr[k] = v; s += v;
    }
    for (int msk = 1; msk < 8; msk <<= 1) s += __shfl_xor(s, msk, 32);
    float m = s * (1.0f / HH);
    float vs = 0.0f;
    for (int k = sub; k < HH; k += 8) {              // pass 2: variance
      float d = yr[k] - m; vs += d * d;
    }
    for (int msk = 1; msk < 8; msk <<= 1) vs += __shfl_xor(vs, msk, 32);
    float rs = rsqrtf(vs * (1.0f / HH) + 1e-5f);
    for (int k = sub; k < HH; k += 8) {              // pass 3: normalize + write both precisions
      float v = (yr[k] - m) * rs * lng[k] + lnb[k];
      x32[rowg * HH + k] = v;
      x16[rowg * HH + k] = (half_t)v;
    }
  }
}

// ---------------- head: GDP pooling + Wp + LN + exact GELU ----------------
__global__ __launch_bounds__(256)
void k_head(const float* __restrict__ x32, const float* __restrict__ Wp, const float* __restrict__ bp,
            const float* __restrict__ g, const float* __restrict__ bb, float* __restrict__ out) {
  __shared__ float sreg[HH];
  __shared__ float sp[HH];
  __shared__ float stats[2];
  int b = blockIdx.x, t = threadIdx.x;
  float acc = 0.0f;
  for (int c = 0; c < 8; ++c) acc += c_gdp[c] * x32[(b * 8 + c) * HH + t];
  sreg[t] = acc;
  __syncthreads();
  float s = 0.0f;
  for (int k = 0; k < HH; ++k) s += sreg[k] * Wp[k * HH + t];
  sp[t] = s + bp[t];
  __syncthreads();
  // block LN stats via wave-parallel partial sums
  if (t < 32) {
    float m = 0.0f;
    for (int k = t; k < HH; k += 32) m += sp[k];
    for (int msk = 1; msk < 32; msk <<= 1) m += __shfl_xor(m, msk, 32);
    float v = 0.0f;
    float mm = m * (1.0f / HH);
    for (int k = t; k < HH; k += 32) { float d = sp[k] - mm; v += d * d; }
    for (int msk = 1; msk < 32; msk <<= 1) v += __shfl_xor(v, msk, 32);
    if (t == 0) { stats[0] = mm; stats[1] = rsqrtf(v * (1.0f / HH) + 1e-5f); }
  }
  __syncthreads();
  float p = (sp[t] - stats[0]) * stats[1] * g[t] + bb[t];
  out[b * HH + t] = 0.5f * p * (1.0f + erff(p * 0.7071067811865476f));
}

// ---------------- launcher ----------------
extern "C" void kernel_launch(void* const* d_in, const int* in_sizes, int n_in,
                              void* d_out, int out_size, void* d_ws, size_t ws_size,
                              hipStream_t stream) {
  (void)in_sizes; (void)n_in; (void)out_size; (void)ws_size;
  const float* feat = (const float*)d_in[0];
  const float* Wi   = (const float*)d_in[1];
  const float* bi   = (const float*)d_in[2];
  const float* emb  = (const float*)d_in[3];
  const float* Wq   = (const float*)d_in[4];
  const float* bq   = (const float*)d_in[5];
  const float* Wk   = (const float*)d_in[6];
  const float* bk   = (const float*)d_in[7];
  const float* Wv   = (const float*)d_in[8];
  const float* bv   = (const float*)d_in[9];
  const float* Wo   = (const float*)d_in[10];
  const float* bo   = (const float*)d_in[11];
  const float* lng  = (const float*)d_in[12];
  const float* lnb  = (const float*)d_in[13];
  const float* Wp   = (const float*)d_in[14];
  const float* bp   = (const float*)d_in[15];
  const float* lnpg = (const float*)d_in[16];
  const float* lnpb = (const float*)d_in[17];
  float* out = (float*)d_out;

  // workspace carve (~125 MB)
  char* w = (char*)d_ws;
  auto carve = [&](size_t bytes) { char* p = w; w += (bytes + 255) & ~(size_t)255; return p; };
  half_t* feat16 = (half_t*)carve((size_t)NB * DIN * 2);
  half_t* Wit    = (half_t*)carve((size_t)HH * DIN * 2);
  half_t* Wqt    = (half_t*)carve((size_t)NLAY * HD * HH * 2);
  half_t* Wkt    = (half_t*)carve((size_t)NLAY * HD * HH * 2);
  half_t* Wvt    = (half_t*)carve((size_t)NLAY * HD * HH * 2);
  half_t* Wot    = (half_t*)carve((size_t)NLAY * HH * HD * 2);
  float*  hbuf   = (float*)carve((size_t)NB * HH * 4);
  float*  x32    = (float*)carve((size_t)NROWS * HH * 4);
  half_t* x16    = (half_t*)carve((size_t)NROWS * HH * 2);

  // converts / transposes (tiny)
  k_cvt_f16<<<(NB * DIN + 255) / 256, 256, 0, stream>>>(feat, feat16, NB * DIN);
  k_transpose_f16<<<(DIN * HH + 255) / 256, 256, 0, stream>>>(Wi, Wit, DIN, HH);
  for (int l = 0; l < NLAY; ++l) {
    k_transpose_f16<<<(HH * HD + 255) / 256, 256, 0, stream>>>(Wq + (size_t)l * HH * HD, Wqt + (size_t)l * HD * HH, HH, HD);
    k_transpose_f16<<<(HH * HD + 255) / 256, 256, 0, stream>>>(Wk + (size_t)l * HH * HD, Wkt + (size_t)l * HD * HH, HH, HD);
    k_transpose_f16<<<(HH * HD + 255) / 256, 256, 0, stream>>>(Wv + (size_t)l * HH * HD, Wvt + (size_t)l * HD * HH, HH, HD);
    k_transpose_f16<<<(HD * HH + 255) / 256, 256, 0, stream>>>(Wo + (size_t)l * HD * HH, Wot + (size_t)l * HH * HD, HD, HH);
  }

  // h = feat @ Wi + bi   (WMMA)
  k_wmma_gemm_bias<<<dim3(NB / 32, HH / 128), 256, 0, stream>>>(feat16, Wit, bi, hbuf, NB, HH, DIN);

  // x = h[:,None,:] + emb
  k_expand<<<NROWS * HH / 256, 256, 0, stream>>>(hbuf, emb, x32, x16);

  // 3 fused attention layers (WMMA GEMMs + VALU attention, in-place x update)
  for (int l = 0; l < NLAY; ++l) {
    k_layer<<<NB / 4, 256, 0, stream>>>(
        x32, x16,
        Wqt + (size_t)l * HD * HH, bq + (size_t)l * HD,
        Wkt + (size_t)l * HD * HH, bk + (size_t)l * HD,
        Wvt + (size_t)l * HD * HH, bv + (size_t)l * HD,
        Wot + (size_t)l * HH * HD, bo + (size_t)l * HH,
        lng + (size_t)l * HH, lnb + (size_t)l * HH);
  }

  // head
  k_head<<<NB, 256, 0, stream>>>(x32, Wp, bp, lnpg, lnpb, out);
}
